// M2RAGNN_82446192214704
// MI455X (gfx1250) — matrix operand order repaired and checked by reading, not compiled, not run on hardware
//
#include <hip/hip_runtime.h>
#include <hip/hip_bf16.h>

// ---------------------------------------------------------------------------
// CDNA5 (gfx1250) implementation of the M2R-AGNN reference pipeline.
// All GEMM-shaped math runs through v_wmma_f32_16x16x32_bf16 (wave32 WMMA).
// Weight fragments are staged in LDS; A fragments preloaded to registers so
// WMMAs issue back-to-back instead of stalling on global loads.
// ---------------------------------------------------------------------------

typedef __bf16 bf16_t;
typedef __attribute__((ext_vector_type(16))) __bf16 v16bf;
typedef __attribute__((ext_vector_type(8)))  __bf16 v8bf;
typedef __attribute__((ext_vector_type(8)))  float  v8f;

#define HDIM 128
#define FDIM 64

// ---- A-fragment loader: 16-bit A 16x32 layout (ISA 7.12.2) ----------------
// lane L: row = base + (L&15); kbase = (L<16 ? 0 : 8)
//   elems 0..7  -> K = k0 + kbase + j        (contiguous 8)
//   elems 8..15 -> K = k0 + kbase + 16 + j-8 (contiguous 8)
__device__ __forceinline__ v16bf load_a32(const bf16_t* rowp, int k0, int lane) {
    int kb = k0 + ((lane & 16) ? 8 : 0);
    v8bf lo = *(const v8bf*)(rowp + kb);
    v8bf hi = *(const v8bf*)(rowp + kb + 16);
    v16bf r;
#pragma unroll
    for (int i = 0; i < 8; ++i) { r[i] = lo[i]; r[i + 8] = hi[i]; }
    return r;
}

// ---- weight pre-pack into B-fragment order --------------------------------
// B 32x16 (16-bit): lane L: col = nt*16 + (L&15); K = kt*32 + (L<16?0:8) + (j<8 ? j : 16+j-8)
// packed[((kt*ntiles + nt)*32 + lane)*16 + j]
__global__ void pack_b_kernel(const float* __restrict__ W, bf16_t* __restrict__ out,
                              int K, int Ncol) {
    int t = blockIdx.x * blockDim.x + threadIdx.x;
    int total = K * Ncol;
    if (t >= total) return;
    int j    = t & 15;
    int lane = (t >> 4) & 31;
    int tile = t >> 9;
    int ntiles = Ncol >> 4;
    int kt = tile / ntiles, nt = tile % ntiles;
    int col = nt * 16 + (lane & 15);
    int k   = kt * 32 + ((lane & 16) ? 8 : 0) + (j < 8 ? j : 16 + (j - 8));
    out[t] = (bf16_t)W[k * Ncol + col];
}

// ---- f32 -> bf16 conversion ------------------------------------------------
__global__ void cvt_bf16_kernel(const float* __restrict__ x, bf16_t* __restrict__ y, int n) {
    for (int i = blockIdx.x * blockDim.x + threadIdx.x; i < n; i += gridDim.x * blockDim.x)
        y[i] = (bf16_t)x[i];
}

__global__ void zero_kernel(float* __restrict__ x, int n) {
    for (int i = blockIdx.x * blockDim.x + threadIdx.x; i < n; i += gridDim.x * blockDim.x)
        x[i] = 0.0f;
}

// ---- generic GEMM: C[M,NCOL] = A_bf[M,K] @ B_packed + bias ----------------
// one wave per 16x16 output tile; A/B fragments preloaded, then an unbroken
// WMMA chain. K, NCOL compile-time for constant addressing.
template <int K, int NCOL>
__global__ void gemm_bias_kernel(const bf16_t* __restrict__ A, int M,
                                 const bf16_t* __restrict__ Bp,
                                 const float* __restrict__ bias,
                                 float* __restrict__ Cf, bf16_t* __restrict__ Cbf) {
    constexpr int KT = K / 32;
    constexpr int NTILES = NCOL / 16;
    int wid  = (blockIdx.x * blockDim.x + threadIdx.x) >> 5;
    int lane = threadIdx.x & 31;
    int mtiles = M >> 4;
    if (wid >= mtiles * NTILES) return;
    int mt = wid / NTILES, nt = wid % NTILES;

    const bf16_t* arow = A + (size_t)(mt * 16 + (lane & 15)) * K;
    v16bf af[KT], bf[KT];
#pragma unroll
    for (int kt = 0; kt < KT; ++kt)
        af[kt] = load_a32(arow, kt * 32, lane);
#pragma unroll
    for (int kt = 0; kt < KT; ++kt)
        bf[kt] = *(const v16bf*)(Bp + ((size_t)(kt * NTILES + nt) * 32 + lane) * 16);

    v8f acc = {0.f, 0.f, 0.f, 0.f, 0.f, 0.f, 0.f, 0.f};
#pragma unroll
    for (int kt = 0; kt < KT; ++kt)
        acc = __builtin_amdgcn_wmma_f32_16x16x32_bf16(false, af[kt], false, bf[kt],
                                                      (short)0, acc, false, false);

    int col  = nt * 16 + (lane & 15);
    float bv = bias ? bias[col] : 0.f;
    int row0 = mt * 16 + ((lane & 16) ? 8 : 0);
#pragma unroll
    for (int v = 0; v < 8; ++v) {
        float val = acc[v] + bv;
        size_t idx = (size_t)(row0 + v) * NCOL + col;
        if (Cf)  Cf[idx]  = val;
        if (Cbf) Cbf[idx] = (bf16_t)val;
    }
}

// ---- edge attention MLP: 16 edges per wave --------------------------------
// s = relu([x_i, x_j, dist] @ W_att1 + b_att1) @ W_att2 + b_att2
// Both packed weight halves (64 KB bf16) staged in LDS per workgroup.
__global__ void edge_mlp_kernel(const bf16_t* __restrict__ h_bf, const float* __restrict__ pos,
                                const int* __restrict__ srcI, const int* __restrict__ dstI,
                                const bf16_t* __restrict__ W1i_p, const bf16_t* __restrict__ W1j_p,
                                const float* __restrict__ W_att1, const float* __restrict__ b_att1,
                                const float* __restrict__ W_att2, const float* __restrict__ b_att2,
                                float* __restrict__ scores, int E) {
    __shared__ bf16_t smem[2 * HDIM * HDIM];   // 64 KB (WGP has 320 KB LDS)
    {
        const uint4* gi = (const uint4*)W1i_p;
        const uint4* gj = (const uint4*)W1j_p;
        uint4* s = (uint4*)smem;
        for (int i = threadIdx.x; i < 2048; i += blockDim.x) s[i] = gi[i];
        for (int i = threadIdx.x; i < 2048; i += blockDim.x) s[2048 + i] = gj[i];
    }
    __syncthreads();
    const bf16_t* w1i = smem;
    const bf16_t* w1j = smem + HDIM * HDIM;

    int wid  = (blockIdx.x * blockDim.x + threadIdx.x) >> 5;
    int lane = threadIdx.x & 31;
    int tiles = (E + 15) >> 4;
    if (wid >= tiles) return;

    int ecol = wid * 16 + (lane & 15);
    if (ecol >= E) ecol = E - 1;
    const bf16_t* arow_i = h_bf + (size_t)dstI[ecol] * HDIM;  // x_i
    const bf16_t* arow_j = h_bf + (size_t)srcI[ecol] * HDIM;  // x_j

    // preload all A fragments (one load-wait, then pure WMMA + DS traffic)
    v16bf ai[4], aj[4];
#pragma unroll
    for (int kt = 0; kt < 4; ++kt) ai[kt] = load_a32(arow_i, kt * 32, lane);
#pragma unroll
    for (int kt = 0; kt < 4; ++kt) aj[kt] = load_a32(arow_j, kt * 32, lane);

    v8f zero = {0.f, 0.f, 0.f, 0.f, 0.f, 0.f, 0.f, 0.f};
    v8f acc[8];
#pragma unroll
    for (int nt = 0; nt < 8; ++nt) acc[nt] = zero;

#pragma unroll
    for (int kt = 0; kt < 4; ++kt) {
#pragma unroll
        for (int nt = 0; nt < 8; ++nt) {
            v16bf b = *(const v16bf*)(w1i + ((size_t)(kt * 8 + nt) * 32 + lane) * 16);
            acc[nt] = __builtin_amdgcn_wmma_f32_16x16x32_bf16(false, ai[kt], false, b,
                                                              (short)0, acc[nt], false, false);
        }
    }
#pragma unroll
    for (int kt = 0; kt < 4; ++kt) {
#pragma unroll
        for (int nt = 0; nt < 8; ++nt) {
            v16bf b = *(const v16bf*)(w1j + ((size_t)(kt * 8 + nt) * 32 + lane) * 16);
            acc[nt] = __builtin_amdgcn_wmma_f32_16x16x32_bf16(false, aj[kt], false, b,
                                                              (short)0, acc[nt], false, false);
        }
    }

    // dist columns (rows 256..258 of W_att1), bias, relu, second layer dot
    int row0 = wid * 16 + ((lane & 16) ? 8 : 0);
    float dx[8], dy[8], dz[8];
#pragma unroll
    for (int v = 0; v < 8; ++v) {
        int e = row0 + v; if (e >= E) e = E - 1;
        int s_ = srcI[e], d_ = dstI[e];
        dx[v] = pos[d_ * 3 + 0] - pos[s_ * 3 + 0];
        dy[v] = pos[d_ * 3 + 1] - pos[s_ * 3 + 1];
        dz[v] = pos[d_ * 3 + 2] - pos[s_ * 3 + 2];
    }
    float partial[8] = {0, 0, 0, 0, 0, 0, 0, 0};
    int cb = lane & 15;
#pragma unroll
    for (int nt = 0; nt < 8; ++nt) {
        int c = nt * 16 + cb;
        float w0 = W_att1[256 * HDIM + c];
        float w1 = W_att1[257 * HDIM + c];
        float w2 = W_att1[258 * HDIM + c];
        float bb = b_att1[c];
        float wo = W_att2[c];
#pragma unroll
        for (int v = 0; v < 8; ++v) {
            float t = acc[nt][v] + dx[v] * w0 + dy[v] * w1 + dz[v] * w2 + bb;
            partial[v] += fmaxf(t, 0.f) * wo;
        }
    }
#pragma unroll
    for (int m = 1; m < 16; m <<= 1)
#pragma unroll
        for (int v = 0; v < 8; ++v) partial[v] += __shfl_xor(partial[v], m, 32);

    if (cb == 0) {
        float bo = b_att2[0];
#pragma unroll
        for (int v = 0; v < 8; ++v) {
            int e = row0 + v;
            if (e < E) scores[e] = partial[v] + bo;
        }
    }
}

// ---- global softmax reductions --------------------------------------------
__global__ void reduce_max_part(const float* __restrict__ s, int n, float* __restrict__ part) {
    __shared__ float sm[256];
    float m = -3.4e38f;
    for (int i = blockIdx.x * blockDim.x + threadIdx.x; i < n; i += gridDim.x * blockDim.x)
        m = fmaxf(m, s[i]);
    sm[threadIdx.x] = m; __syncthreads();
    for (int o = 128; o > 0; o >>= 1) {
        if ((int)threadIdx.x < o) sm[threadIdx.x] = fmaxf(sm[threadIdx.x], sm[threadIdx.x + o]);
        __syncthreads();
    }
    if (threadIdx.x == 0) part[blockIdx.x] = sm[0];
}
__global__ void reduce_max_final(const float* __restrict__ part, int n, float* __restrict__ out) {
    __shared__ float sm[256];
    float m = -3.4e38f;
    for (int i = threadIdx.x; i < n; i += 256) m = fmaxf(m, part[i]);
    sm[threadIdx.x] = m; __syncthreads();
    for (int o = 128; o > 0; o >>= 1) {
        if ((int)threadIdx.x < o) sm[threadIdx.x] = fmaxf(sm[threadIdx.x], sm[threadIdx.x + o]);
        __syncthreads();
    }
    if (threadIdx.x == 0) out[0] = sm[0];
}
__global__ void reduce_sum_part(const float* __restrict__ s, int n,
                                const float* __restrict__ smax, float* __restrict__ part) {
    __shared__ float sm[256];
    float mx = smax[0];
    float acc = 0.f;
    for (int i = blockIdx.x * blockDim.x + threadIdx.x; i < n; i += gridDim.x * blockDim.x)
        acc += expf(s[i] - mx);
    sm[threadIdx.x] = acc; __syncthreads();
    for (int o = 128; o > 0; o >>= 1) {
        if ((int)threadIdx.x < o) sm[threadIdx.x] += sm[threadIdx.x + o];
        __syncthreads();
    }
    if (threadIdx.x == 0) part[blockIdx.x] = sm[0];
}
__global__ void reduce_sum_final(const float* __restrict__ part, int n, float* __restrict__ out_inv) {
    __shared__ float sm[256];
    float acc = 0.f;
    for (int i = threadIdx.x; i < n; i += 256) acc += part[i];
    sm[threadIdx.x] = acc; __syncthreads();
    for (int o = 128; o > 0; o >>= 1) {
        if ((int)threadIdx.x < o) sm[threadIdx.x] += sm[threadIdx.x + o];
        __syncthreads();
    }
    if (threadIdx.x == 0) out_inv[0] = 1.0f / sm[0];
}

// ---- weighted message scatter-add: aggr[dst] += h[src] * att --------------
__global__ void scatter_kernel(const float* __restrict__ h, const float* __restrict__ scores,
                               const int* __restrict__ srcI, const int* __restrict__ dstI,
                               const float* __restrict__ smax, const float* __restrict__ inv_sum,
                               float* __restrict__ aggr, int E) {
    long tid = (long)blockIdx.x * blockDim.x + threadIdx.x;
    long total = (long)E * 32;
    if (tid >= total) return;
    int e = (int)(tid >> 5);
    int part = (int)(tid & 31);
    float att = expf(scores[e] - smax[0]) * inv_sum[0];
    const float4 v = *(const float4*)(h + (size_t)srcI[e] * HDIM + part * 4);
    float* dp = aggr + (size_t)dstI[e] * HDIM + part * 4;
    atomicAdd(dp + 0, v.x * att);
    atomicAdd(dp + 1, v.y * att);
    atomicAdd(dp + 2, v.z * att);
    atomicAdd(dp + 3, v.w * att);
}

// ---- fused head: out = relu(A@W1 + b1) @ W2 + b2  (K=128, hidden=64) ------
__global__ void head_mlp_kernel(const bf16_t* __restrict__ A, int M,
                                const bf16_t* __restrict__ W1p, const float* __restrict__ b1,
                                const float* __restrict__ W2, const float* __restrict__ b2,
                                float* __restrict__ out) {
    __shared__ bf16_t smem[HDIM * 64];   // 16 KB
    {
        const uint4* g = (const uint4*)W1p;
        uint4* s = (uint4*)smem;
        for (int i = threadIdx.x; i < 1024; i += blockDim.x) s[i] = g[i];
    }
    __syncthreads();

    int wid  = (blockIdx.x * blockDim.x + threadIdx.x) >> 5;
    int lane = threadIdx.x & 31;
    int tiles = (M + 15) >> 4;
    if (wid >= tiles) return;

    int rcol = wid * 16 + (lane & 15);
    if (rcol >= M) rcol = M - 1;
    const bf16_t* arow = A + (size_t)rcol * HDIM;

    v16bf af[4];
#pragma unroll
    for (int kt = 0; kt < 4; ++kt) af[kt] = load_a32(arow, kt * 32, lane);

    v8f zero = {0.f, 0.f, 0.f, 0.f, 0.f, 0.f, 0.f, 0.f};
    v8f acc[4];
#pragma unroll
    for (int nt = 0; nt < 4; ++nt) acc[nt] = zero;

#pragma unroll
    for (int kt = 0; kt < 4; ++kt) {
#pragma unroll
        for (int nt = 0; nt < 4; ++nt) {
            v16bf b = *(const v16bf*)(smem + ((size_t)(kt * 4 + nt) * 32 + lane) * 16);
            acc[nt] = __builtin_amdgcn_wmma_f32_16x16x32_bf16(false, af[kt], false, b,
                                                              (short)0, acc[nt], false, false);
        }
    }
    float partial[8] = {0, 0, 0, 0, 0, 0, 0, 0};
    int cb = lane & 15;
#pragma unroll
    for (int nt = 0; nt < 4; ++nt) {
        int c = nt * 16 + cb;
        float bb = b1[c], wo = W2[c];
#pragma unroll
        for (int v = 0; v < 8; ++v)
            partial[v] += fmaxf(acc[nt][v] + bb, 0.f) * wo;
    }
#pragma unroll
    for (int m = 1; m < 16; m <<= 1)
#pragma unroll
        for (int v = 0; v < 8; ++v) partial[v] += __shfl_xor(partial[v], m, 32);

    if (cb == 0) {
        int row0 = wid * 16 + ((lane & 16) ? 8 : 0);
        float bo = b2[0];
#pragma unroll
        for (int v = 0; v < 8; ++v) {
            int r = row0 + v;
            if (r < M) out[r] = partial[v] + bo;
        }
    }
}

// ---------------------------------------------------------------------------
extern "C" void kernel_launch(void* const* d_in, const int* in_sizes, int n_in,
                              void* d_out, int out_size, void* d_ws, size_t ws_size,
                              hipStream_t stream) {
    const float* mol_x      = (const float*)d_in[0];
    const float* pos        = (const float*)d_in[1];
    const float* reaction_x = (const float*)d_in[2];
    const float* target_x   = (const float*)d_in[3];
    const int*   edge_index = (const int*)d_in[4];
    const float* W_node = (const float*)d_in[5];
    const float* b_node = (const float*)d_in[6];
    const float* W_att1 = (const float*)d_in[7];
    const float* b_att1 = (const float*)d_in[8];
    const float* W_att2 = (const float*)d_in[9];
    const float* b_att2 = (const float*)d_in[10];
    const float* W_upd  = (const float*)d_in[11];
    const float* b_upd  = (const float*)d_in[12];
    const float* Wy1 = (const float*)d_in[13];
    const float* by1 = (const float*)d_in[14];
    const float* Wy2 = (const float*)d_in[15];
    const float* by2 = (const float*)d_in[16];
    const float* Wa1 = (const float*)d_in[17];
    const float* ba1 = (const float*)d_in[18];
    const float* Wa2 = (const float*)d_in[19];
    const float* ba2 = (const float*)d_in[20];

    const int N  = in_sizes[0] / FDIM;
    const int NR = in_sizes[2] / FDIM;
    const int NT = in_sizes[3] / FDIM;
    const int E  = in_sizes[4] / 2;
    const int* srcI = edge_index;
    const int* dstI = edge_index + E;

    // ---- workspace carving ----
    char* p = (char*)d_ws;
    auto alloc = [&](size_t bytes) -> char* {
        char* r = p; p += (bytes + 255) & ~(size_t)255; return r;
    };
    float*  h_f32   = (float*) alloc((size_t)N * HDIM * 4);
    bf16_t* h_bf    = (bf16_t*)alloc((size_t)N * HDIM * 2);
    bf16_t* mol_bf  = (bf16_t*)alloc((size_t)N * FDIM * 2);
    bf16_t* r_bf    = (bf16_t*)alloc((size_t)NR * FDIM * 2);
    bf16_t* t_bf    = (bf16_t*)alloc((size_t)NT * FDIM * 2);
    bf16_t* rh_bf   = (bf16_t*)alloc((size_t)NR * HDIM * 2);
    bf16_t* th_bf   = (bf16_t*)alloc((size_t)NT * HDIM * 2);
    float*  scores  = (float*) alloc((size_t)E * 4);
    float*  aggr    = (float*) alloc((size_t)N * HDIM * 4);
    bf16_t* aggr_bf = (bf16_t*)alloc((size_t)N * HDIM * 2);
    bf16_t* Wnode_p = (bf16_t*)alloc((size_t)FDIM * HDIM * 2);
    bf16_t* W1i_p   = (bf16_t*)alloc((size_t)HDIM * HDIM * 2);
    bf16_t* W1j_p   = (bf16_t*)alloc((size_t)HDIM * HDIM * 2);
    bf16_t* Wupd_p  = (bf16_t*)alloc((size_t)HDIM * HDIM * 2);
    bf16_t* Wy1_p   = (bf16_t*)alloc((size_t)HDIM * 64 * 2);
    bf16_t* Wa1_p   = (bf16_t*)alloc((size_t)HDIM * 64 * 2);
    float*  partbuf = (float*) alloc(512 * 4);
    float*  smax    = (float*) alloc(256);
    float*  inv_sum = (float*) alloc(256);

    float* out_yield = (float*)d_out;
    float* out_act   = out_yield + NR;
    float* out_feats = out_act + NT;

    const int TB = 256;
    auto cdiv = [](long a, long b) { return (int)((a + b - 1) / b); };

    // 1) pack weights to bf16 B-fragment layouts
    pack_b_kernel<<<cdiv(FDIM * HDIM, TB), TB, 0, stream>>>(W_node, Wnode_p, FDIM, HDIM);
    pack_b_kernel<<<cdiv(HDIM * HDIM, TB), TB, 0, stream>>>(W_att1, W1i_p, HDIM, HDIM);
    pack_b_kernel<<<cdiv(HDIM * HDIM, TB), TB, 0, stream>>>(W_att1 + HDIM * HDIM, W1j_p, HDIM, HDIM);
    pack_b_kernel<<<cdiv(HDIM * HDIM, TB), TB, 0, stream>>>(W_upd, Wupd_p, HDIM, HDIM);
    pack_b_kernel<<<cdiv(HDIM * 64, TB), TB, 0, stream>>>(Wy1, Wy1_p, HDIM, 64);
    pack_b_kernel<<<cdiv(HDIM * 64, TB), TB, 0, stream>>>(Wa1, Wa1_p, HDIM, 64);

    // 2) activations -> bf16
    cvt_bf16_kernel<<<4096, TB, 0, stream>>>(mol_x, mol_bf, N * FDIM);
    cvt_bf16_kernel<<<2048, TB, 0, stream>>>(reaction_x, r_bf, NR * FDIM);
    cvt_bf16_kernel<<<2048, TB, 0, stream>>>(target_x, t_bf, NT * FDIM);

    // 3) node encoder GEMMs (WMMA): h / rh / th
    {
        int waves = (N >> 4) * (HDIM >> 4);
        gemm_bias_kernel<FDIM, HDIM><<<cdiv((long)waves * 32, TB), TB, 0, stream>>>(
            mol_bf, N, Wnode_p, b_node, h_f32, h_bf);
        waves = (NR >> 4) * (HDIM >> 4);
        gemm_bias_kernel<FDIM, HDIM><<<cdiv((long)waves * 32, TB), TB, 0, stream>>>(
            r_bf, NR, Wnode_p, b_node, (float*)nullptr, rh_bf);
        waves = (NT >> 4) * (HDIM >> 4);
        gemm_bias_kernel<FDIM, HDIM><<<cdiv((long)waves * 32, TB), TB, 0, stream>>>(
            t_bf, NT, Wnode_p, b_node, (float*)nullptr, th_bf);
    }

    // 4) edge attention MLP (WMMA-heavy: 64 wmma per wave, weights in LDS)
    {
        int tiles = (E + 15) >> 4;
        edge_mlp_kernel<<<cdiv((long)tiles * 32, TB), TB, 0, stream>>>(
            h_bf, pos, srcI, dstI, W1i_p, W1j_p, W_att1, b_att1, W_att2, b_att2, scores, E);
    }

    // 5) global softmax statistics over all E scores
    reduce_max_part<<<512, TB, 0, stream>>>(scores, E, partbuf);
    reduce_max_final<<<1, TB, 0, stream>>>(partbuf, 512, smax);
    reduce_sum_part<<<512, TB, 0, stream>>>(scores, E, smax, partbuf);
    reduce_sum_final<<<1, TB, 0, stream>>>(partbuf, 512, inv_sum);

    // 6) weighted message scatter-add
    zero_kernel<<<4096, TB, 0, stream>>>(aggr, N * HDIM);
    scatter_kernel<<<cdiv((long)E * 32, TB), TB, 0, stream>>>(
        h_f32, scores, srcI, dstI, smax, inv_sum, aggr, E);

    // 7) update GEMM -> mol_feats (straight into d_out)
    cvt_bf16_kernel<<<4096, TB, 0, stream>>>(aggr, aggr_bf, N * HDIM);
    {
        int waves = (N >> 4) * (HDIM >> 4);
        gemm_bias_kernel<HDIM, HDIM><<<cdiv((long)waves * 32, TB), TB, 0, stream>>>(
            aggr_bf, N, Wupd_p, b_upd, out_feats, (bf16_t*)nullptr);
    }

    // 8) prediction heads (fused WMMA + VALU second layer, weights in LDS)
    {
        int tiles = (NR + 15) >> 4;
        head_mlp_kernel<<<cdiv((long)tiles * 32, TB), TB, 0, stream>>>(
            rh_bf, NR, Wy1_p, by1, Wy2, by2, out_yield);
        tiles = (NT + 15) >> 4;
        head_mlp_kernel<<<cdiv((long)tiles * 32, TB), TB, 0, stream>>>(
            th_bf, NT, Wa1_p, ba1, Wa2, ba2, out_act);
    }
}
